// Head_6734508720397
// MI455X (gfx1250) — matrix operand order repaired and checked by reading, not compiled, not run on hardware
//
#include <hip/hip_runtime.h>
#include <hip/hip_bf16.h>

// CDNA5 / gfx1250 fused attention head (k@q^T variant with causal softmax).
// Pipeline: [prep: transpose+pad weights to f16] -> [QKV projection via
// v_wmma_f32_16x16x32_f16, writing K,Q row-major f16 and V transposed f16]
// -> [flash-style attention, online softmax, WMMA for S and P@V].
// This revision software-pipelines the B-operand loads (double buffering) so
// each WMMA overlaps with the next tile's global_load_b128 pair.

typedef _Float16 half_t;
typedef __attribute__((ext_vector_type(16))) _Float16 v16h;
typedef __attribute__((ext_vector_type(8)))  float    v8f;

#define WMMA_F16(a, b, c) \
  __builtin_amdgcn_wmma_f32_16x16x32_f16(false, (a), false, (b), (short)0, (c), false, false)

constexpr int Bn   = 512;
constexpr int Tn   = 256;
constexpr int EMB  = 364;
constexpr int EMBP = 384;   // EMB padded to multiple of 32 (WMMA K step)
constexpr int HSn  = 64;

__device__ inline v16h load_2x16B(const half_t* p0, const half_t* p1) {
  union { uint4 q[2]; v16h h; } u;
  u.q[0] = *(const uint4*)p0;
  u.q[1] = *(const uint4*)p1;
  return u.h;
}
__device__ inline v16h load_32B(const half_t* p) { return load_2x16B(p, p + 16); }

// ---------------------------------------------------------------------------
// Kernel 0: convert Wk/Wq/Wv (fp32 [EMB][HS] row-major) into transposed,
// K-padded f16 Wt [3][HS][EMBP]. Column-major-in-K so projection B-operand
// fetches are contiguous per lane.
// ---------------------------------------------------------------------------
__global__ void prep_weights(const float* __restrict__ Wk,
                             const float* __restrict__ Wq,
                             const float* __restrict__ Wv,
                             half_t* __restrict__ Wt) {
  int idx = blockIdx.x * 256 + threadIdx.x;
  if (idx >= 3 * HSn * EMBP) return;
  int m = idx / (HSn * EMBP);
  int r = idx % (HSn * EMBP);
  int n = r / EMBP;     // output column (HS)
  int k = r % EMBP;     // input feature
  const float* W = (m == 0) ? Wk : ((m == 1) ? Wq : Wv);
  float v = (k < EMB) ? W[(size_t)k * HSn + n] : 0.0f;
  Wt[idx] = (half_t)v;
}

// ---------------------------------------------------------------------------
// Kernel 1: QKV projection. Block = 256 threads = 8 waves, 128 rows of
// x[B*T, EMB]. Each wave computes a 16-row slab: 12 accumulator tiles
// (3 matrices x 4 column tiles of 16), K-loop of 12 WMMA steps (k=32).
// K,Q stored row-major f16; V stored transposed f16 [B][HS][T].
// ---------------------------------------------------------------------------
__global__ __launch_bounds__(256) void qkv_proj(
    const float* __restrict__ x,
    const float* __restrict__ bk, const float* __restrict__ bq,
    const float* __restrict__ bv,
    const half_t* __restrict__ Wt,
    half_t* __restrict__ kf, half_t* __restrict__ qf,
    half_t* __restrict__ vT) {
  const int lane = threadIdx.x & 31;
  const int wave = threadIdx.x >> 5;
  const int lr   = lane & 15;   // A row / B column selector
  const int hi   = lane >> 4;   // lane-half selector
  const int row0 = blockIdx.x * 128 + wave * 16;   // global row (b*T + t)

  v8f acc[12];
  const float* biases[3] = {bk, bq, bv};
#pragma unroll
  for (int m = 0; m < 3; m++) {
#pragma unroll
    for (int n = 0; n < 4; n++) {
      float bias = biases[m][n * 16 + lr];   // C element col = lr
      v8f c;
#pragma unroll
      for (int j = 0; j < 8; j++) c[j] = bias;
      acc[m * 4 + n] = c;
    }
  }

  const float* xp = x + (size_t)(row0 + lr) * EMB;

  // B tile pointer for output tile i (i = m*4+n) at K-chunk c.
  auto bptr = [&](int i, int c) -> const half_t* {
    int m = i >> 2, n = i & 3;
    return Wt + (size_t)(m * HSn + n * 16 + lr) * EMBP + c * 32 + hi * 8;
  };

  // Double-buffered B operand: load tile i+1 while WMMA consumes tile i.
  auto do_chunk = [&](const v16h& A, int c) {
    v16h bcur = load_32B(bptr(0, c));
#pragma unroll
    for (int i = 0; i < 12; i++) {
      v16h bnext;
      if (i < 11) bnext = load_32B(bptr(i + 1, c));
      acc[i] = WMMA_F16(A, bcur, acc[i]);
      bcur = bnext;
    }
  };

  // Full K chunks (d < 352): vectorized fp32 loads, convert to f16.
#pragma unroll 1
  for (int c = 0; c < 11; c++) {
    const int d0 = c * 32 + hi * 8;
    float4 f0 = *(const float4*)(xp + d0);
    float4 f1 = *(const float4*)(xp + d0 + 4);
    float4 f2 = *(const float4*)(xp + d0 + 16);
    float4 f3 = *(const float4*)(xp + d0 + 20);
    v16h A;
    A[0]=(half_t)f0.x; A[1]=(half_t)f0.y; A[2]=(half_t)f0.z; A[3]=(half_t)f0.w;
    A[4]=(half_t)f1.x; A[5]=(half_t)f1.y; A[6]=(half_t)f1.z; A[7]=(half_t)f1.w;
    A[8]=(half_t)f2.x; A[9]=(half_t)f2.y; A[10]=(half_t)f2.z; A[11]=(half_t)f2.w;
    A[12]=(half_t)f3.x; A[13]=(half_t)f3.y; A[14]=(half_t)f3.z; A[15]=(half_t)f3.w;
    do_chunk(A, c);
  }
  {  // Tail chunk c=11: d in [352, 384), valid only below 364.
    const int c = 11;
    const int d0 = c * 32 + hi * 8;
    v16h A;
#pragma unroll
    for (int e = 0; e < 8; e++) {
      int da = d0 + e, db = d0 + 16 + e;
      A[e]     = (da < EMB) ? (half_t)xp[da] : (half_t)0.0f;
      A[8 + e] = (db < EMB) ? (half_t)xp[db] : (half_t)0.0f;
    }
    do_chunk(A, c);
  }

  // --- epilogue ---
  // K, Q row-major f16. C element (j, lane): row = row0+j+8*hi, col = n*16+lr.
#pragma unroll
  for (int m = 0; m < 2; m++) {
    half_t* dst = (m == 0) ? kf : qf;
#pragma unroll
    for (int n = 0; n < 4; n++) {
      v8f c = acc[m * 4 + n];
#pragma unroll
      for (int j = 0; j < 8; j++) {
        int r = row0 + j + 8 * hi;
        dst[(size_t)r * HSn + n * 16 + lr] = (half_t)c[j];
      }
    }
  }
  // V transposed [B][HS][T]: per lane, fixed d = n*16+lr, t varies over j ->
  // pack 8 halves into one 16B store.
  {
    const int b  = row0 / Tn;
    const int tb = (row0 % Tn) + 8 * hi;
#pragma unroll
    for (int n = 0; n < 4; n++) {
      v8f c = acc[8 + n];
      union { half_t h[8]; uint4 q; } u;
#pragma unroll
      for (int j = 0; j < 8; j++) u.h[j] = (half_t)c[j];
      *(uint4*)(vT + (size_t)(b * HSn + n * 16 + lr) * Tn + tb) = u.q;
    }
  }
}

// ---------------------------------------------------------------------------
// Kernel 2: causal attention, flash-style online softmax.
// Grid (B, 2); block = 8 waves; wave owns a 16-row t-tile (t0 = 128*by+16*w),
// accumulating all HS=64 output columns. s-loop in tiles of 32.
// wei[t,s] = k[t].q[s] * HS^-0.5, masked s<=t.
// ---------------------------------------------------------------------------
__global__ __launch_bounds__(256) void attn(
    const half_t* __restrict__ kf, const half_t* __restrict__ qf,
    const half_t* __restrict__ vT, float* __restrict__ out) {
  __shared__ __align__(16) half_t Pbuf[8][16][40];  // per-wave P bounce, padded

  const int lane = threadIdx.x & 31;
  const int wave = threadIdx.x >> 5;
  const int lr   = lane & 15;
  const int hi   = lane >> 4;
  const int b    = blockIdx.x;
  const int t0   = blockIdx.y * 128 + wave * 16;
  const float NEG_INF = -__builtin_inff();

  // A operand (k rows), loaded once: chunks d=[0,32) and [32,64).
  const half_t* kp = kf + ((size_t)b * Tn + t0 + lr) * HSn;
  const v16h Ak0 = load_2x16B(kp + hi * 8,      kp + hi * 8 + 16);
  const v16h Ak1 = load_2x16B(kp + 32 + hi * 8, kp + 32 + hi * 8 + 16);

  v8f accO[4];
#pragma unroll
  for (int n = 0; n < 4; n++)
#pragma unroll
    for (int j = 0; j < 8; j++) accO[n][j] = 0.0f;
  float mrow[8], lrow[8];
#pragma unroll
  for (int j = 0; j < 8; j++) { mrow[j] = NEG_INF; lrow[j] = 0.0f; }

  half_t (*P)[40] = Pbuf[wave];

  for (int s0 = 0; s0 <= t0; s0 += 32) {
    const bool has1 = (s0 + 16 <= t0);
    float p0v[8], p1v[8];

    {  // sub-tile 0: s in [s0, s0+16)
      const half_t* qp = qf + ((size_t)b * Tn + s0 + lr) * HSn;
      v16h Bq0 = load_32B(qp + hi * 8);        // both chunk loads in flight
      v16h Bq1 = load_32B(qp + 32 + hi * 8);   // before the WMMA chain
      v8f z;
#pragma unroll
      for (int j = 0; j < 8; j++) z[j] = 0.0f;
      v8f c = WMMA_F16(Ak0, Bq0, z);
      c     = WMMA_F16(Ak1, Bq1, c);
#pragma unroll
      for (int j = 0; j < 8; j++) {
        float s = c[j] * 0.125f;                       // HS^-0.5
        if (s0 == t0 && lr > j + 8 * hi) s = NEG_INF;  // diagonal mask
        p0v[j] = s;
      }
    }
    if (has1) {  // sub-tile 1: s in [s0+16, s0+32)
      const half_t* qp = qf + ((size_t)b * Tn + s0 + 16 + lr) * HSn;
      v16h Bq0 = load_32B(qp + hi * 8);
      v16h Bq1 = load_32B(qp + 32 + hi * 8);
      v8f z;
#pragma unroll
      for (int j = 0; j < 8; j++) z[j] = 0.0f;
      v8f c = WMMA_F16(Ak0, Bq0, z);
      c     = WMMA_F16(Ak1, Bq1, c);
#pragma unroll
      for (int j = 0; j < 8; j++) {
        float s = c[j] * 0.125f;
        if (s0 + 16 == t0 && lr > j + 8 * hi) s = NEG_INF;
        p1v[j] = s;
      }
    } else {
#pragma unroll
      for (int j = 0; j < 8; j++) p1v[j] = NEG_INF;
    }

    // Online softmax per row (row data lives across 16 lanes of a half-wave).
    float corr[8];
#pragma unroll
    for (int j = 0; j < 8; j++) {
      float rm = fmaxf(p0v[j], p1v[j]);
#pragma unroll
      for (int off = 8; off >= 1; off >>= 1)
        rm = fmaxf(rm, __shfl_xor(rm, off, 32));
      float mn = fmaxf(mrow[j], rm);
      float cf = __expf(mrow[j] - mn);
      mrow[j]  = mn;
      float e0 = __expf(p0v[j] - mn);
      float e1 = __expf(p1v[j] - mn);
      p0v[j] = e0; p1v[j] = e1;
      float rs = e0 + e1;
#pragma unroll
      for (int off = 8; off >= 1; off >>= 1) rs += __shfl_xor(rs, off, 32);
      lrow[j] = lrow[j] * cf + rs;
      corr[j] = cf;
    }
#pragma unroll
    for (int n = 0; n < 4; n++)
#pragma unroll
      for (int j = 0; j < 8; j++) accO[n][j] *= corr[j];

    // P (C-layout) -> LDS -> A-layout. Same-wave DS ops are in-order; add an
    // explicit dscnt wait before the cross-lane reads.
#pragma unroll
    for (int j = 0; j < 8; j++) {
      int t = j + 8 * hi;
      P[t][lr]      = (half_t)p0v[j];
      P[t][16 + lr] = (half_t)p1v[j];
    }
    asm volatile("s_wait_dscnt 0" ::: "memory");
    v16h Ap = load_2x16B(&P[lr][hi * 8], &P[lr][hi * 8 + 16]);

    // O += P @ V : B[k=s][n=d] = vT[d][s], two 16B loads per lane, 32B apart.
    // Double-buffered so the next tile's load overlaps the current WMMA.
    {
      auto vptr = [&](int n) -> const half_t* {
        return vT + (size_t)(b * HSn + n * 16 + lr) * Tn + s0 + hi * 8;
      };
      v16h Bv = load_32B(vptr(0));
#pragma unroll
      for (int n = 0; n < 4; n++) {
        v16h Bnext;
        if (n < 3) Bnext = load_32B(vptr(n + 1));
        accO[n] = WMMA_F16(Ap, Bv, accO[n]);
        Bv = Bnext;
      }
    }
  }

  // Epilogue: out[t][d] = accO / l, fp32 row-major.
#pragma unroll
  for (int n = 0; n < 4; n++) {
#pragma unroll
    for (int j = 0; j < 8; j++) {
      int t = t0 + j + 8 * hi;
      out[((size_t)b * Tn + t) * HSn + n * 16 + lr] = accO[n][j] / lrow[j];
    }
  }
}

// ---------------------------------------------------------------------------
extern "C" void kernel_launch(void* const* d_in, const int* in_sizes, int n_in,
                              void* d_out, int out_size, void* d_ws,
                              size_t ws_size, hipStream_t stream) {
  const float* x  = (const float*)d_in[0];
  const float* Wk = (const float*)d_in[1];
  const float* bk = (const float*)d_in[2];
  const float* Wq = (const float*)d_in[3];
  const float* bq = (const float*)d_in[4];
  const float* Wv = (const float*)d_in[5];
  const float* bv = (const float*)d_in[6];
  float* out = (float*)d_out;

  // Workspace layout (f16): kf, qf row-major [B*T, HS]; vT [B][HS][T]; Wt.
  const size_t Nkqv = (size_t)Bn * Tn * HSn;            // 8,388,608 halves each
  char* ws = (char*)d_ws;
  half_t* kf = (half_t*)ws;
  half_t* qf = kf + Nkqv;
  half_t* vT = qf + Nkqv;
  half_t* Wt = vT + Nkqv;                               // 3*HS*EMBP halves

  {
    int total = 3 * HSn * EMBP;
    prep_weights<<<(total + 255) / 256, 256, 0, stream>>>(Wk, Wq, Wv, Wt);
  }
  {
    int blocks = (Bn * Tn) / 128;                        // 1024
    qkv_proj<<<blocks, 256, 0, stream>>>(x, bk, bq, bv, Wt, kf, qf, vT);
  }
  {
    dim3 grid(Bn, Tn / 128);                             // (512, 2)
    attn<<<grid, 256, 0, stream>>>(kf, qf, vT, out);
  }
}
